// GRU_Model3_47837345743488
// MI455X (gfx1250) — matrix operand order repaired and checked by reading, not compiled, not run on hardware
//
#include <hip/hip_runtime.h>

// ---------------------------------------------------------------------------
// GRU (reset_after=True) on gfx1250 with v_wmma_f32_16x16x32_bf16.
//   1) pack_weights : fp32 weights -> bf16 WMMA B-fragment order
//   2) xproj        : x @ kernel + bias_i (bf16 WMMA), stored bf16 in WMMA
//                     C-fragment order [T][mt][nt]
//   3) 128x gru_step: hp = h @ rkernel + bias_r via WMMA, gates fused.
//                     Two-deep software pipeline over K (ping-pong fragment
//                     sets) because this kernel runs ~1 wave/SIMD and must
//                     hide L2 latency with per-wave ILP.
//   4) dense head
// Workspace (~208 MB): Wk_pack | Wr_pack | xp_pack | hA | hB | hA_bf | hB_bf
// ---------------------------------------------------------------------------

#define B_  512
#define T_  128
#define F_  512
#define H_  512
#define G3_ 1536

typedef __bf16 bf16_t;
typedef bf16_t v16bf __attribute__((ext_vector_type(16)));
typedef float  v8f   __attribute__((ext_vector_type(8)));

union Frag16 {
    v16bf v;
    unsigned short u[16];
    uint4 q[2];
};

__device__ __forceinline__ unsigned short f2bf(float f) {
    unsigned int u = __float_as_uint(f);
    u += 0x7FFFu + ((u >> 16) & 1u);      // round-to-nearest-even
    return (unsigned short)(u >> 16);
}
__device__ __forceinline__ float bf2f(unsigned short s) {
    return __uint_as_float(((unsigned int)s) << 16);
}
__device__ __forceinline__ float sigmoid_f(float x) {
    return 1.0f / (1.0f + __expf(-x));
}
__device__ __forceinline__ float tanh_f(float x) {
    float xc = fminf(fmaxf(x, -15.0f), 15.0f);
    float e  = __expf(-2.0f * xc);
    return (1.0f - e) / (1.0f + e);
}

// ---------------------------------------------------------------------------
// 1) Pack fp32 [K,3H] weights into bf16 WMMA B-fragment order.
//    B tile = 32(K) x 16(N). Lane l holds column n=l%16, h=l/16;
//    element e<8: K=8h+e, e>=8: K=16+8h+(e-8). Tile = 32 lanes * 32B = 1KB.
//    Tile order: [nt][kt], tile offset (nt*16 + kt).
// ---------------------------------------------------------------------------
__global__ __launch_bounds__(256) void pack_weights_kernel(
    const float* __restrict__ wk, const float* __restrict__ wr,
    unsigned short* __restrict__ wk_pack, unsigned short* __restrict__ wr_pack)
{
    int idx = blockIdx.x * 256 + threadIdx.x;
    const int total = F_ * G3_;
    if (idx >= 2 * total) return;
    const float* src = (idx < total) ? wk : wr;
    unsigned short* dst = (idx < total) ? wk_pack : wr_pack;
    int e = idx % total;
    int n = e % G3_;
    int k = e / G3_;
    int lane = (n & 15) + (((k >> 3) & 1) << 4);   // +16 if (k%16)>=8
    int elem = (k & 7) + (((k >> 4) & 1) << 3);
    int kt = k >> 5;                               // K-chunk of 32
    int nt = n >> 4;
    size_t off = ((size_t)(nt * (F_ / 32) + kt)) * 512 + (size_t)lane * 16 + elem;
    dst[off] = f2bf(src[(size_t)k * G3_ + n]);
}

// ---------------------------------------------------------------------------
// 2) x_proj = x @ kernel + bias_i, stored bf16 in C-fragment order:
//    xp_pack[((t*32+mt)*96+nt)*256 + lane*8 + v]  (ushort units; 512B/tile)
//    Each wave: one 16-row M tile (rows b = mt*16+m, fixed t), 8 N tiles.
//    Grid: t(128) * mt(32) * nchunk(12) = 49152 waves = 6144 blocks x 8 waves.
//    (Massively parallel: cross-wave occupancy hides the per-wmma waits.)
// ---------------------------------------------------------------------------
__global__ __launch_bounds__(256) void xproj_kernel(
    const float* __restrict__ x,
    const unsigned short* __restrict__ wk_pack,
    const float* __restrict__ bias_i,
    unsigned short* __restrict__ xp_pack)
{
    const int lane = threadIdx.x & 31;
    const int wave = threadIdx.x >> 5;
    int task = blockIdx.x * 8 + wave;
    const int nc = task % 12;  task /= 12;
    const int mt = task % 32;  task /= 32;
    const int t  = task;                     // 0..127
    const int m  = lane & 15;
    const int hi = lane >> 4;

    v8f acc[8] = {};

    const int b = mt * 16 + m;
    const float* xrow = x + ((size_t)b * T_ + t) * F_;

    for (int kt = 0; kt < 16; ++kt) {
        // A source floats + all 8 B fragments
        const float* ap = xrow + kt * 32 + hi * 8;
        float4 f0 = *(const float4*)(ap);
        float4 f1 = *(const float4*)(ap + 4);
        float4 f2 = *(const float4*)(ap + 16);
        float4 f3 = *(const float4*)(ap + 20);

        Frag16 bfr[8];
#pragma unroll
        for (int ni = 0; ni < 8; ++ni) {
            const int nt = nc * 8 + ni;
            const unsigned short* bp =
                wk_pack + ((size_t)(nt * 16 + kt)) * 512 + (size_t)lane * 16;
            bfr[ni].q[0] = *(const uint4*)(bp);
            bfr[ni].q[1] = *(const uint4*)(bp + 8);
        }

        // A fragment 16x32 bf16: lane row m, K = kt*32 + hi*8 + {0..7,16..23}
        Frag16 a;
        a.u[0]  = f2bf(f0.x); a.u[1]  = f2bf(f0.y); a.u[2]  = f2bf(f0.z); a.u[3]  = f2bf(f0.w);
        a.u[4]  = f2bf(f1.x); a.u[5]  = f2bf(f1.y); a.u[6]  = f2bf(f1.z); a.u[7]  = f2bf(f1.w);
        a.u[8]  = f2bf(f2.x); a.u[9]  = f2bf(f2.y); a.u[10] = f2bf(f2.z); a.u[11] = f2bf(f2.w);
        a.u[12] = f2bf(f3.x); a.u[13] = f2bf(f3.y); a.u[14] = f2bf(f3.z); a.u[15] = f2bf(f3.w);

#pragma unroll
        for (int ni = 0; ni < 8; ++ni) {
            acc[ni] = __builtin_amdgcn_wmma_f32_16x16x32_bf16(
                false, a.v, false, bfr[ni].v, (short)0, acc[ni], false, false);
        }
    }

#pragma unroll
    for (int ni = 0; ni < 8; ++ni) {
        const int nt = nc * 8 + ni;
        const float bia = bias_i[nt * 16 + m];   // column = lane%16, same for all v
        unsigned int w[4];
#pragma unroll
        for (int p = 0; p < 4; ++p) {
            float lo = acc[ni][2 * p]     + bia;
            float hv = acc[ni][2 * p + 1] + bia;
            w[p] = (unsigned int)f2bf(lo) | ((unsigned int)f2bf(hv) << 16);
        }
        unsigned short* dst =
            xp_pack + (((size_t)t * 32 + mt) * 96 + nt) * 256 + (size_t)lane * 8;
        uint4 q; q.x = w[0]; q.y = w[1]; q.z = w[2]; q.w = w[3];
        *(uint4*)dst = q;
    }
}

// ---------------------------------------------------------------------------
// 3) One GRU time step, software-pipelined.
//    Wave = 2 M tiles (32 rows) x 1 N tile (16 cols), 3 band accumulators
//    (z / r / h) sharing each A fragment.
//    Grid: mt2(16) * nt(32) = 512 waves = 64 blocks x 8 waves.
// ---------------------------------------------------------------------------
struct StepFrags {
    Frag16 bz, br, bh;   // weight fragments for z/r/h bands
    Frag16 a[2];         // h-state fragments for the 2 M tiles
};

__device__ __forceinline__ void load_step_frags(
    StepFrags& f,
    const unsigned short* __restrict__ wr_pack,
    const unsigned short* __restrict__ h_in_bf,
    int nt, int kt, int lane, int m, int hi, int mt2)
{
    const unsigned short* p0 =
        wr_pack + ((size_t)((nt)      * 16 + kt)) * 512 + (size_t)lane * 16;
    const unsigned short* p1 =
        wr_pack + ((size_t)((nt + 32) * 16 + kt)) * 512 + (size_t)lane * 16;
    const unsigned short* p2 =
        wr_pack + ((size_t)((nt + 64) * 16 + kt)) * 512 + (size_t)lane * 16;
    f.bz.q[0] = *(const uint4*)(p0); f.bz.q[1] = *(const uint4*)(p0 + 8);
    f.br.q[0] = *(const uint4*)(p1); f.br.q[1] = *(const uint4*)(p1 + 8);
    f.bh.q[0] = *(const uint4*)(p2); f.bh.q[1] = *(const uint4*)(p2 + 8);
#pragma unroll
    for (int mi = 0; mi < 2; ++mi) {
        const int b = mt2 * 32 + mi * 16 + m;
        const unsigned short* ap = h_in_bf + (size_t)b * H_ + kt * 32 + hi * 8;
        f.a[mi].q[0] = *(const uint4*)(ap);
        f.a[mi].q[1] = *(const uint4*)(ap + 16);
    }
}

__device__ __forceinline__ void wmma_step(v8f acc[2][3], const StepFrags& f)
{
#pragma unroll
    for (int mi = 0; mi < 2; ++mi) {
        acc[mi][0] = __builtin_amdgcn_wmma_f32_16x16x32_bf16(
            false, f.a[mi].v, false, f.bz.v, (short)0, acc[mi][0], false, false);
        acc[mi][1] = __builtin_amdgcn_wmma_f32_16x16x32_bf16(
            false, f.a[mi].v, false, f.br.v, (short)0, acc[mi][1], false, false);
        acc[mi][2] = __builtin_amdgcn_wmma_f32_16x16x32_bf16(
            false, f.a[mi].v, false, f.bh.v, (short)0, acc[mi][2], false, false);
    }
}

__global__ __launch_bounds__(256)
__attribute__((amdgpu_waves_per_eu(1)))
void gru_step_kernel(
    const unsigned short* __restrict__ xp_pack,
    const unsigned short* __restrict__ wr_pack,
    const float* __restrict__ bias_r,
    const float* __restrict__ h_in,               // fp32 [B,H]
    const unsigned short* __restrict__ h_in_bf,   // bf16 [B,H]
    float* __restrict__ h_out,
    unsigned short* __restrict__ h_out_bf,
    int t)
{
    const int lane = threadIdx.x & 31;
    const int wave = threadIdx.x >> 5;
    const int task = blockIdx.x * 8 + wave;       // 0..511
    const int nt   = task & 31;
    const int mt2  = task >> 5;                   // 0..15
    const int m    = lane & 15;
    const int hi   = lane >> 4;

    v8f acc[2][3] = {};

    // ---- two-deep software pipeline over the 16 K-chunks -------------------
    StepFrags f0, f1;
    load_step_frags(f0, wr_pack, h_in_bf, nt, 0, lane, m, hi, mt2);
#pragma unroll
    for (int kt = 0; kt < 16; kt += 2) {
        load_step_frags(f1, wr_pack, h_in_bf, nt, kt + 1, lane, m, hi, mt2);
        wmma_step(acc, f0);                       // overlaps f1's loads
        if (kt + 2 < 16) {
            load_step_frags(f0, wr_pack, h_in_bf, nt, kt + 2, lane, m, hi, mt2);
        }
        wmma_step(acc, f1);                       // overlaps f0's loads
    }

    const int n_g = nt * 16 + m;
    const float bzs = bias_r[n_g];
    const float brs = bias_r[H_ + n_g];
    const float bhs = bias_r[2 * H_ + n_g];

#pragma unroll
    for (int mi = 0; mi < 2; ++mi) {
        const int mt = mt2 * 2 + mi;
        const size_t tb = ((size_t)t * 32 + mt) * 96;
        uint4 qz = *(const uint4*)(xp_pack + (tb + nt)      * 256 + (size_t)lane * 8);
        uint4 qr = *(const uint4*)(xp_pack + (tb + nt + 32) * 256 + (size_t)lane * 8);
        uint4 qh = *(const uint4*)(xp_pack + (tb + nt + 64) * 256 + (size_t)lane * 8);
        float fz[8], fr[8], fh[8];
        {
            unsigned int wz[4]  = {qz.x, qz.y, qz.z, qz.w};
            unsigned int wr2[4] = {qr.x, qr.y, qr.z, qr.w};
            unsigned int wh[4]  = {qh.x, qh.y, qh.z, qh.w};
#pragma unroll
            for (int p = 0; p < 4; ++p) {
                fz[2 * p]     = bf2f((unsigned short)(wz[p] & 0xFFFFu));
                fz[2 * p + 1] = bf2f((unsigned short)(wz[p] >> 16));
                fr[2 * p]     = bf2f((unsigned short)(wr2[p] & 0xFFFFu));
                fr[2 * p + 1] = bf2f((unsigned short)(wr2[p] >> 16));
                fh[2 * p]     = bf2f((unsigned short)(wh[p] & 0xFFFFu));
                fh[2 * p + 1] = bf2f((unsigned short)(wh[p] >> 16));
            }
        }
#pragma unroll
        for (int v = 0; v < 8; ++v) {
            const int row = mt * 16 + v + 8 * hi;     // C-layout: M = v + 8*(lane>=16)
            const size_t idx = (size_t)row * H_ + n_g;
            float hold = h_in[idx];
            float z  = sigmoid_f(fz[v] + acc[mi][0][v] + bzs);
            float r  = sigmoid_f(fr[v] + acc[mi][1][v] + brs);
            float hh = tanh_f(fh[v] + r * (acc[mi][2][v] + bhs));
            float hn = z * hold + (1.0f - z) * hh;
            h_out[idx]    = hn;
            h_out_bf[idx] = f2bf(hn);
        }
    }
}

// ---------------------------------------------------------------------------
// h0 = 0 (both ping-pong buffers, fp32 + bf16 shadows)
// ---------------------------------------------------------------------------
__global__ __launch_bounds__(256) void zero_h_kernel(
    float* hA, float* hB, unsigned short* hAb, unsigned short* hBb)
{
    int i = blockIdx.x * 256 + threadIdx.x;
    if (i < B_ * H_) {
        hA[i] = 0.0f; hB[i] = 0.0f; hAb[i] = 0; hBb[i] = 0;
    }
}

// ---------------------------------------------------------------------------
// 4) Dense head: out[b] = h_last[b,:] @ dense_w + dense_b
// ---------------------------------------------------------------------------
__global__ __launch_bounds__(256) void dense_kernel(
    const float* __restrict__ h, const float* __restrict__ w,
    const float* __restrict__ bias, float* __restrict__ out)
{
    int r = blockIdx.x * 256 + threadIdx.x;
    if (r >= B_) return;
    const float* hr = h + (size_t)r * H_;
    float s = 0.0f;
    for (int k = 0; k < H_; k += 4) {
        float4 hv = *(const float4*)(hr + k);
        s += hv.x * w[k] + hv.y * w[k + 1] + hv.z * w[k + 2] + hv.w * w[k + 3];
    }
    out[r] = s + bias[0];
}

// ---------------------------------------------------------------------------
extern "C" void kernel_launch(void* const* d_in, const int* in_sizes, int n_in,
                              void* d_out, int out_size, void* d_ws, size_t ws_size,
                              hipStream_t stream)
{
    (void)in_sizes; (void)n_in; (void)out_size; (void)ws_size;
    const float* x       = (const float*)d_in[0];
    const float* kern    = (const float*)d_in[1];
    const float* rkern   = (const float*)d_in[2];
    const float* bias_i  = (const float*)d_in[3];
    const float* bias_r  = (const float*)d_in[4];
    const float* dense_w = (const float*)d_in[5];
    const float* dense_b = (const float*)d_in[6];
    float* out = (float*)d_out;

    // Workspace layout (needs ~208 MB)
    char* ws = (char*)d_ws;
    unsigned short* wk_pack = (unsigned short*)(ws);                       // 1.5 MB
    unsigned short* wr_pack = (unsigned short*)(ws + (((size_t)2) << 20)); // 1.5 MB
    unsigned short* xp_pack = (unsigned short*)(ws + (((size_t)4) << 20)); // 192 MB
    const size_t xp_bytes = (size_t)T_ * 32 * 96 * 512;                    // 201326592
    char* p = ws + (((size_t)4) << 20) + xp_bytes;
    float* hA = (float*)p;              p += (size_t)B_ * H_ * 4;
    float* hB = (float*)p;              p += (size_t)B_ * H_ * 4;
    unsigned short* hAb = (unsigned short*)p; p += (size_t)B_ * H_ * 2;
    unsigned short* hBb = (unsigned short*)p;

    pack_weights_kernel<<<(2 * F_ * G3_ + 255) / 256, 256, 0, stream>>>(
        kern, rkern, wk_pack, wr_pack);

    xproj_kernel<<<6144, 256, 0, stream>>>(x, wk_pack, bias_i, xp_pack);

    zero_h_kernel<<<(B_ * H_ + 255) / 256, 256, 0, stream>>>(hA, hB, hAb, hBb);

    for (int t = 0; t < T_; ++t) {
        const bool even = (t & 1) == 0;
        gru_step_kernel<<<64, 256, 0, stream>>>(
            xp_pack, wr_pack, bias_r,
            even ? hA : hB, even ? hAb : hBb,
            even ? hB : hA, even ? hBb : hAb, t);
    }
    // T_=128 (even number of swaps) -> final state back in hA
    dense_kernel<<<(B_ + 255) / 256, 256, 0, stream>>>(hA, dense_w, dense_b, out);
}
